// GAT_70076686402255
// MI455X (gfx1250) — matrix operand order repaired and checked by reading, not compiled, not run on hardware
//
#include <hip/hip_runtime.h>
#include <hip/hip_bf16.h>
#include <stdint.h>

// ---------------------------------------------------------------------------
// GAT forward for MI455X (gfx1250, wave32). GEMMs use v_wmma_f32_16x16x32_f16,
// fragment-major packed weights, register blocking 5 M-tiles x 2 N-tiles/wave
// (M=50000 = 625*80; Ncols in {128,256} = multiples of 128).
// ---------------------------------------------------------------------------

typedef __attribute__((ext_vector_type(16))) _Float16 v16h;
typedef __attribute__((ext_vector_type(8)))  float    v8f;

#define HIDC 128   // per-head channel count (constant in this model)
#define MT   5     // M-tiles (16 rows each) per wave: 50000 = 625 * (16*5)
#define NT   2     // N-tiles (16 cols each) per wave

// ---------------- utility kernels ----------------

__global__ void k_f32_to_f16(const float* __restrict__ src, _Float16* __restrict__ dst, int n) {
  int i = blockIdx.x * blockDim.x + threadIdx.x;
  if (i < n) dst[i] = (_Float16)src[i];
}

__global__ void k_fill(float* __restrict__ p, float val, int n) {
  int i = blockIdx.x * blockDim.x + threadIdx.x;
  if (i < n) p[i] = val;
}

// Pack weight B[K,Ncols] (f32) into fragment-major f16:
// fragment (kt,nt) = 32x16 tile; lane L holds 16 contiguous f16:
//   elem i  <-  B[(kt*32 + (L>>4)*16 + i) * Ncols + nt*16 + (L&15)]
// Packed index: (((kt*(Ncols/16)) + nt) * 32 + L) * 16 + i
__global__ void k_pack_w(const float* __restrict__ W, _Float16* __restrict__ P,
                         int K, int Ncols) {
  int t = blockIdx.x * blockDim.x + threadIdx.x;
  if (t >= K * Ncols) return;
  int i      = t & 15;
  int L      = (t >> 4) & 31;
  int frag   = t >> 9;
  int ntiles = Ncols >> 4;
  int nt = frag % ntiles;
  int kt = frag / ntiles;
  int k = kt * 32 + (L >> 4) * 16 + i;
  int n = nt * 16 + (L & 15);
  P[t] = (_Float16)W[(size_t)k * Ncols + n];
}

// ---------------- WMMA GEMM ----------------
// C[M,Ncols] = A[M,K](f16, row major) x Bpacked[K,Ncols](fragment major)
// Block = 128 threads = 4 waves; wave owns MT=5 M-tiles x NT=2 N-tiles.
// grid.x = Ncols/128, grid.y = M/80. Exact division -> EXEC all-ones.
// Per k-step: 5 A-fragment + 2 B-fragment loads feed 10 independent WMMAs.

__global__ void __launch_bounds__(128)
k_gemm_wmma(const _Float16* __restrict__ A, const _Float16* __restrict__ Bp,
            const float* __restrict__ bias, float* __restrict__ C,
            _Float16* __restrict__ C16, int M, int K, int Ncols, int relu)
{
  const int wave = threadIdx.x >> 5;
  const int lane = threadIdx.x & 31;
  const int half = lane >> 4;     // 0: lanes 0-15, 1: lanes 16-31
  const int l15  = lane & 15;
  const int nt0  = (blockIdx.x * 4 + wave) * NT;   // first N-tile of this wave
  const int m0   = blockIdx.y * (16 * MT);

  v8f acc[MT][NT];
#pragma unroll
  for (int u = 0; u < NT; ++u) {
    const float bv = bias ? bias[(nt0 + u) * 16 + l15] : 0.0f;
#pragma unroll
    for (int t = 0; t < MT; ++t)
#pragma unroll
      for (int r = 0; r < 8; ++r) acc[t][u][r] = bv;
  }

  const int ntiles = Ncols >> 4;
  const v16h* __restrict__ Bf = (const v16h*)Bp;
  // A fragment: lane holds row M = l15 (both halves), two contiguous 16B chunks:
  //   av[0..7]  = A[m, k0 + half*16 + 0..7]
  //   av[8..15] = A[m, k0 + half*16 + 8 + 0..7]
  const _Float16* __restrict__ Abase = A + (size_t)(m0 + l15) * K;

  for (int k0 = 0; k0 < K; k0 += 32) {
    v16h bfrag[NT];
#pragma unroll
    for (int u = 0; u < NT; ++u)
      bfrag[u] = Bf[((size_t)(k0 >> 5) * ntiles + nt0 + u) * 32 + lane];
#pragma unroll
    for (int t = 0; t < MT; ++t) {
      const _Float16* Ar = Abase + (size_t)t * 16 * K + k0 + half * 16;
      v16h av;
#pragma unroll
      for (int j = 0; j < 8; ++j) av[j] = Ar[j];
#pragma unroll
      for (int j = 0; j < 8; ++j) av[8 + j] = Ar[8 + j];
#pragma unroll
      for (int u = 0; u < NT; ++u)
        acc[t][u] = __builtin_amdgcn_wmma_f32_16x16x32_f16(
            /*neg_a=*/false, av, /*neg_b=*/false, bfrag[u],
            /*c_mod=*/(short)0, acc[t][u], /*reuse_a=*/false, /*reuse_b=*/false);
    }
  }

  // C/D layout: lane -> N = l15 ; VGPR r -> M = r + 8*half
#pragma unroll
  for (int t = 0; t < MT; ++t) {
#pragma unroll
    for (int u = 0; u < NT; ++u) {
#pragma unroll
      for (int r = 0; r < 8; ++r) {
        float val = acc[t][u][r];
        if (relu && val < 0.0f) val = 0.0f;
        size_t idx = (size_t)(m0 + t * 16 + r + half * 8) * Ncols + (nt0 + u) * 16 + l15;
        if (C)   C[idx] = val;
        if (C16) C16[idx] = (_Float16)val;
      }
    }
  }
}

// ---------------- attention coefficients: s/d = <h[n,hd,:], a[hd,:]> ----------------
// One wave per (node, head); HIDC=128 = 32 lanes x float4, then shuffle reduce.

__global__ void k_attn_coef(const float* __restrict__ Hf, const float* __restrict__ asrc,
                            const float* __restrict__ adst, float* __restrict__ S,
                            float* __restrict__ D, int Nn, int Hh)
{
  int wid  = (int)((blockIdx.x * blockDim.x + threadIdx.x) >> 5);
  int lane = threadIdx.x & 31;
  if (wid >= Nn * Hh) return;
  int hd = wid % Hh;
  const float4* hp = (const float4*)(Hf + (size_t)wid * HIDC);  // wid*HIDC == n*Hh*HIDC + hd*HIDC
  const float4* ap = (const float4*)(asrc + hd * HIDC);
  const float4* bp = (const float4*)(adst + hd * HIDC);
  float4 h4 = hp[lane];
  float4 a4 = ap[lane];
  float4 b4 = bp[lane];
  float ss = h4.x * a4.x + h4.y * a4.y + h4.z * a4.z + h4.w * a4.w;
  float dd = h4.x * b4.x + h4.y * b4.y + h4.z * b4.z + h4.w * b4.w;
  for (int o = 16; o; o >>= 1) {
    ss += __shfl_down(ss, o, 32);
    dd += __shfl_down(dd, o, 32);
  }
  if (lane == 0) { S[wid] = ss; D[wid] = dd; }
}

// ---------------- edge kernels (segment softmax over dst + aggregation) ----------------

__device__ __forceinline__ void atomicMaxF(float* addr, float val) {
  if (__float_as_int(val) >= 0)
    atomicMax((int*)addr, __float_as_int(val));
  else
    atomicMin((unsigned int*)addr, __float_as_uint(val));
}

__device__ __forceinline__ float leaky02(float v) { return v > 0.0f ? v : 0.2f * v; }

// Pass 1: compute leaky scores (stored to EX) + segment max over dst.
__global__ void k_edge_max(const int* __restrict__ ei, int E, int ET,
                           const float* __restrict__ S, const float* __restrict__ D,
                           float* __restrict__ Mx, float* __restrict__ EX, int Hh)
{
  int e = blockIdx.x * blockDim.x + threadIdx.x;
  if (e >= ET) return;
  int s, d;
  if (e < E) { s = ei[e]; d = ei[E + e]; } else { s = e - E; d = s; }
  for (int hd = 0; hd < Hh; ++hd) {
    float v = leaky02(S[s * Hh + hd] + D[d * Hh + hd]);
    EX[(size_t)e * Hh + hd] = v;
    atomicMaxF(&Mx[d * Hh + hd], v);
  }
}

// Pass 2: EX <- exp(score - max[dst]); accumulate denom.
__global__ void k_edge_expsum(const int* __restrict__ ei, int E, int ET,
                              const float* __restrict__ Mx, float* __restrict__ EX,
                              float* __restrict__ Dn, int Hh)
{
  int e = blockIdx.x * blockDim.x + threadIdx.x;
  if (e >= ET) return;
  int d = (e < E) ? ei[E + e] : e - E;
  for (int hd = 0; hd < Hh; ++hd) {
    float ex = expf(EX[(size_t)e * Hh + hd] - Mx[d * Hh + hd]);
    EX[(size_t)e * Hh + hd] = ex;
    atomicAdd(&Dn[d * Hh + hd], ex);
  }
}

// Pass 3: one wave per edge: lanes stride the Hh*HIDC features, f32 atomic scatter.
__global__ void k_edge_agg(const int* __restrict__ ei, int E, int ET,
                           const float* __restrict__ EX, const float* __restrict__ Dn,
                           const float* __restrict__ Hf, float* __restrict__ OUT, int Hh)
{
  int wid  = (int)((blockIdx.x * blockDim.x + threadIdx.x) >> 5);
  int lane = threadIdx.x & 31;
  if (wid >= ET) return;
  int s, d;
  if (wid < E) { s = ei[wid]; d = ei[E + wid]; } else { s = wid - E; d = s; }
  float a0 = EX[(size_t)wid * Hh + 0] / Dn[d * Hh + 0];
  float a1 = (Hh > 1) ? EX[(size_t)wid * Hh + 1] / Dn[d * Hh + 1] : a0;
  const int HC = Hh * HIDC;
  const float* hs = Hf + (size_t)s * HC;
  float*       op = OUT + (size_t)d * HC;
  for (int i = lane; i < HC; i += 32) {
    float al = (i < HIDC) ? a0 : a1;
    atomicAdd(&op[i], al * hs[i]);
  }
}

// ---------------- finalize: add bias, emit f16 activation and/or f32 output ----------------

__global__ void k_finalize(const float* __restrict__ OUT, const float* __restrict__ bias,
                           _Float16* __restrict__ a16, float* __restrict__ fout,
                           int total, int HCmask)
{
  int i = blockIdx.x * blockDim.x + threadIdx.x;
  if (i >= total) return;
  float v = OUT[i] + bias[i & HCmask];   // HC is a power of two
  if (a16)  a16[i]  = (_Float16)v;
  if (fout) fout[i] = v;
}

// ---------------------------------------------------------------------------
// host orchestration
// ---------------------------------------------------------------------------

extern "C" void kernel_launch(void* const* d_in, const int* in_sizes, int n_in,
                              void* d_out, int out_size, void* d_ws, size_t ws_size,
                              hipStream_t stream)
{
  const float* x     = (const float*)d_in[0];
  const int*   ei    = (const int*)d_in[1];   // [2, E]
  const float* fc_w  = (const float*)d_in[2];
  const float* fc_b  = (const float*)d_in[3];
  const float* W1    = (const float*)d_in[4];
  const float* asrc1 = (const float*)d_in[5];
  const float* adst1 = (const float*)d_in[6];
  const float* b1    = (const float*)d_in[7];
  const float* Wm    = (const float*)d_in[8];   // [4,256,256]
  const float* asrcm = (const float*)d_in[9];   // [4,2,128]
  const float* adstm = (const float*)d_in[10];  // [4,2,128]
  const float* bm    = (const float*)d_in[11];  // [4,256]
  const float* W6    = (const float*)d_in[12];
  const float* asrc6 = (const float*)d_in[13];
  const float* adst6 = (const float*)d_in[14];
  const float* b6    = (const float*)d_in[15];

  const int Nn = in_sizes[0] / 64;   // 50000 nodes
  const int E  = in_sizes[1] / 2;    // 800000 edges
  const int ET = E + Nn;             // + self loops

  // ---- workspace carve-out (256B aligned) ----
  size_t off = 0;
  auto wsAlloc = [&](size_t bytes) -> void* {
    void* p = (char*)d_ws + off;
    off += (bytes + 255) & ~(size_t)255;
    return p;
  };
  _Float16* bufA  = (_Float16*)wsAlloc((size_t)Nn * 64  * sizeof(_Float16)); // x in f16
  _Float16* bufB  = (_Float16*)wsAlloc((size_t)Nn * 256 * sizeof(_Float16)); // activations f16
  float*    Hf    = (float*)wsAlloc((size_t)Nn * 256 * sizeof(float));       // h = x@W
  float*    OUTb  = (float*)wsAlloc((size_t)Nn * 256 * sizeof(float));       // aggregation
  float*    Sb    = (float*)wsAlloc((size_t)Nn * 2 * sizeof(float));
  float*    Db    = (float*)wsAlloc((size_t)Nn * 2 * sizeof(float));
  float*    Mx    = (float*)wsAlloc((size_t)Nn * 2 * sizeof(float));
  float*    Dn    = (float*)wsAlloc((size_t)Nn * 2 * sizeof(float));
  float*    EXb   = (float*)wsAlloc((size_t)ET * 2 * sizeof(float));
  _Float16* fcw16 = (_Float16*)wsAlloc((size_t)64 * 128 * sizeof(_Float16));     // packed
  _Float16* w1_16 = (_Float16*)wsAlloc((size_t)128 * 256 * sizeof(_Float16));    // packed
  _Float16* wm16  = (_Float16*)wsAlloc((size_t)4 * 256 * 256 * sizeof(_Float16));// packed
  _Float16* w6_16 = (_Float16*)wsAlloc((size_t)256 * 128 * sizeof(_Float16));    // packed
  (void)ws_size; (void)n_in; (void)out_size;

  const int TB = 256;
  auto cdiv = [](int a, int b) { return (a + b - 1) / b; };

  // ---- input conversion + weight packing (fragment-major f16) ----
  k_f32_to_f16<<<cdiv(Nn * 64, TB), TB, 0, stream>>>(x, bufA, Nn * 64);
  k_pack_w<<<cdiv(64 * 128, TB), TB, 0, stream>>>(fc_w, fcw16, 64, 128);
  k_pack_w<<<cdiv(128 * 256, TB), TB, 0, stream>>>(W1, w1_16, 128, 256);
  for (int i = 0; i < 4; ++i)
    k_pack_w<<<cdiv(256 * 256, TB), TB, 0, stream>>>(Wm + (size_t)i * 256 * 256,
                                                     wm16 + (size_t)i * 256 * 256, 256, 256);
  k_pack_w<<<cdiv(256 * 128, TB), TB, 0, stream>>>(W6, w6_16, 256, 128);

  // ---- FC layer: relu(x @ fc_w + fc_b) -> bufB (f16) ----
  {
    dim3 g(128 / (64 * NT / 2) / 2 + (128 / 128) - 1, 1, 1); // = Ncols/128 = 1
    dim3 grid(128 / 128, Nn / (16 * MT));
    k_gemm_wmma<<<grid, 128, 0, stream>>>(bufA, fcw16, fc_b, (float*)nullptr, bufB,
                                          Nn, 64, 128, 1);
  }

  // ---- one GAT layer ----
  auto run_gat = [&](const _Float16* Ain, const _Float16* Wl, const float* asrc,
                     const float* adst, const float* bias, int Kdim, int Hh,
                     _Float16* a16out, float* fout) {
    const int Ncols = Hh * HIDC;
    dim3 grid(Ncols / 128, Nn / (16 * MT));
    k_gemm_wmma<<<grid, 128, 0, stream>>>(Ain, Wl, (const float*)nullptr, Hf,
                                          (_Float16*)nullptr, Nn, Kdim, Ncols, 0);
    int nwv = Nn * Hh;  // one wave per (node, head)
    k_attn_coef<<<cdiv(nwv, 8), TB, 0, stream>>>(Hf, asrc, adst, Sb, Db, Nn, Hh);
    k_fill<<<cdiv(Nn * Hh, TB), TB, 0, stream>>>(Mx, -1e30f, Nn * Hh);
    k_fill<<<cdiv(Nn * Hh, TB), TB, 0, stream>>>(Dn, 0.0f, Nn * Hh);
    k_fill<<<cdiv(Nn * Ncols, TB), TB, 0, stream>>>(OUTb, 0.0f, Nn * Ncols);
    k_edge_max<<<cdiv(ET, TB), TB, 0, stream>>>(ei, E, ET, Sb, Db, Mx, EXb, Hh);
    k_edge_expsum<<<cdiv(ET, TB), TB, 0, stream>>>(ei, E, ET, Mx, EXb, Dn, Hh);
    k_edge_agg<<<cdiv(ET, 8), TB, 0, stream>>>(ei, E, ET, EXb, Dn, Hf, OUTb, Hh);
    k_finalize<<<cdiv(Nn * Ncols, TB), TB, 0, stream>>>(OUTb, bias, a16out, fout,
                                                        Nn * Ncols, Ncols - 1);
  };

  // layer 1: [N,128] -> [N,256], heads=2
  run_gat(bufB, w1_16, asrc1, adst1, b1, 128, 2, bufB, (float*)nullptr);
  // layers 2-5: [N,256] -> [N,256], heads=2
  for (int i = 0; i < 4; ++i) {
    run_gat(bufB, wm16 + (size_t)i * 256 * 256, asrcm + (size_t)i * 2 * HIDC,
            adstm + (size_t)i * 2 * HIDC, bm + (size_t)i * 256, 256, 2, bufB,
            (float*)nullptr);
  }
  // layer 6: [N,256] -> [N,128], heads=1, writes final f32 output
  run_gat(bufB, w6_16, asrc6, adst6, b6, 256, 1, (_Float16*)nullptr, (float*)d_out);
}
